// Contrastive_D_74294344286288
// MI455X (gfx1250) — compile-verified, matrix operand order checked by reading, not run on hardware
//
#include <hip/hip_runtime.h>
#include <hip/hip_bf16.h>
#include <math.h>

typedef __attribute__((ext_vector_type(16))) __bf16 v16bf;
typedef __attribute__((ext_vector_type(8)))  __bf16 v8bf;
typedef __attribute__((ext_vector_type(8)))  float  v8f;
typedef __attribute__((ext_vector_type(4)))  int    v4i;

#define WN    8192
#define WBITS 128
#define WM    16
#define WC    100
#define INVT  (1.0f/0.07f)
#define LAMW  0.5f

// workspace layout (float offsets)
#define OFF_LSE   0
#define OFF_CNT   8192
#define OFF_MEAN  8320
#define OFF_SEG   12416
#define OFF_ZEND  (12416 + WC*WM*256)   // end of zero-init region = 422016
#define OFF_BHI   422016                 // bf16 hi plane of b: N*128 bf16 = 524288 float slots
#define OFF_BLO   (OFF_BHI + 262144*2)   // 946304
#define OFF_W2T   (OFF_BLO + 262144*2)   // 1470592 (W2t: M*256*256 bf16)

#define AS1 __attribute__((address_space(1)))
#define AS3 __attribute__((address_space(3)))

#if __has_builtin(__builtin_amdgcn_global_load_async_to_lds_b128)
#define HAVE_ASYNC 1
#else
#define HAVE_ASYNC 0
#endif

// CDNA5 16-bit A/B fragment K base for this lane: lanes 0-15 -> K in {0..7,16..23},
// lanes 16-31 -> K in {8..15,24..31}. Both runs are 8 contiguous elements.
__device__ __forceinline__ int koff_of(int lane) { return (lane < 16) ? 0 : 8; }

// Build a v16bf fragment from two contiguous 8-element runs.
__device__ __forceinline__ v16bf frag_from(const __bf16* base) {
  v8bf lo = *(const v8bf*)(base);        // K = koff + 0..7
  v8bf hi = *(const v8bf*)(base + 16);   // K = koff + 16..23
  return __builtin_shufflevector(lo, hi, 0,1,2,3,4,5,6,7,8,9,10,11,12,13,14,15);
}

// 16-byte global -> LDS copy: CDNA5 async DMA when available, sync fallback otherwise.
__device__ __forceinline__ void copy16_g2l(const __bf16* g, __bf16* l) {
#if HAVE_ASYNC
  __builtin_amdgcn_global_load_async_to_lds_b128((AS1 v4i*)g, (AS3 v4i*)l, 0, 0);
#else
  *(v8bf*)l = *(const v8bf*)g;
#endif
}
#if HAVE_ASYNC
__device__ __forceinline__ void wait_async_2() {
#if __has_builtin(__builtin_amdgcn_s_wait_asynccnt)
  __builtin_amdgcn_s_wait_asynccnt(2);
#else
  asm volatile("s_wait_asynccnt 0x2" ::: "memory");
#endif
}
__device__ __forceinline__ void wait_async_0() {
#if __has_builtin(__builtin_amdgcn_s_wait_asynccnt)
  __builtin_amdgcn_s_wait_asynccnt(0);
#else
  asm volatile("s_wait_asynccnt 0x0" ::: "memory");
#endif
}
#endif

// ---------------- init: zero accumulators + output ----------------
__global__ void __launch_bounds__(256) init_kernel(float* ws, float* out) {
  int idx = blockIdx.x * 256 + threadIdx.x;
  if (idx == 0) out[0] = 0.0f;
  for (int i = OFF_CNT + idx; i < OFF_ZEND; i += gridDim.x * 256) ws[i] = 0.0f;
}

// ---------------- prep: b -> bf16 hi/lo planes (bf16x3 split done once) ----------------
__global__ void __launch_bounds__(256) prep_b_kernel(const float* __restrict__ b,
                                                     __bf16* __restrict__ bhi,
                                                     __bf16* __restrict__ blo) {
  int i = blockIdx.x * 256 + threadIdx.x;
  if (i >= WN * WBITS) return;
  float f = b[i];
  __bf16 h = (__bf16)f;
  bhi[i] = h;
  blo[i] = (__bf16)(f - (float)h);
}

// ---------------- prep: W2 -> transposed bf16 W2t[m][o][k] ----------------
__global__ void __launch_bounds__(256) prep_w2_kernel(const float* __restrict__ W2,
                                                      __bf16* __restrict__ w2t) {
  int i = blockIdx.x * 256 + threadIdx.x;
  if (i >= WM * 256 * 256) return;
  int m = i >> 16, r = i & 65535, o = r >> 8, k = r & 255;
  w2t[m * 65536 + o * 256 + k] = (__bf16)W2[m * 65536 + k * 256 + o];
}

// ---------------- kernel 1: per-row logsumexp of b@b^T/T (excl. diag) ----------------
// 8 waves x 16 rows; A fragments (hi/lo) loaded straight from bf16 planes; j-tile
// double-buffered into LDS with async DMA; bf16x3 WMMA; zero cvt in the hot loop.
__global__ void __launch_bounds__(256) lse_kernel(const __bf16* __restrict__ bhi,
                                                  const __bf16* __restrict__ blo,
                                                  float* __restrict__ lse) {
  __shared__ __bf16 bufHi[2][16 * WBITS];  // 4 KB each
  __shared__ __bf16 bufLo[2][16 * WBITS];
  const int tid  = threadIdx.x;
  const int lane = tid & 31;
  const int wave = tid >> 5;
  const int rowTile = blockIdx.x * 128 + wave * 16;
  const int nrow = lane & 15;
  const int arow = rowTile + nrow;
  const int ko = koff_of(lane);

  // A fragments (hi/lo), K = 128 -> 4 slices of 32
  v16bf ahi[4], alo[4];
#pragma unroll
  for (int s = 0; s < 4; ++s) {
    ahi[s] = frag_from(bhi + arow * WBITS + 32 * s + ko);
    alo[s] = frag_from(blo + arow * WBITS + 32 * s + ko);
  }

  float mrow[8], srow[8];
#pragma unroll
  for (int r = 0; r < 8; ++r) { mrow[r] = -1e30f; srow[r] = 0.0f; }
  const int crow0 = rowTile + ((lane < 16) ? 0 : 8);

  const int NT = WN / 16;
  // prologue: stage tile 0 (each thread moves one 16B chunk of hi and of lo)
  copy16_g2l(bhi + 0 * 16 * WBITS + tid * 8, &bufHi[0][tid * 8]);
  copy16_g2l(blo + 0 * 16 * WBITS + tid * 8, &bufLo[0][tid * 8]);

  for (int jt = 0; jt < NT; ++jt) {
    const int cur = jt & 1;
    __syncthreads();  // everyone done reading buf[1-cur] from iteration jt-1
    if (jt + 1 < NT) {
      copy16_g2l(bhi + (jt + 1) * 16 * WBITS + tid * 8, &bufHi[1 - cur][tid * 8]);
      copy16_g2l(blo + (jt + 1) * 16 * WBITS + tid * 8, &bufLo[1 - cur][tid * 8]);
    }
#if HAVE_ASYNC
    if (jt + 1 < NT) wait_async_2(); else wait_async_0();  // drain tile jt
#endif
    __syncthreads();  // tile jt fully in LDS, WG-wide

    v8f c = {};
#pragma unroll
    for (int s = 0; s < 4; ++s) {
      v16bf bh = frag_from(&bufHi[cur][nrow * WBITS + 32 * s + ko]);
      v16bf bl = frag_from(&bufLo[cur][nrow * WBITS + 32 * s + ko]);
      // bf16x3: hi*hi + hi*lo + lo*hi
      c = __builtin_amdgcn_wmma_f32_16x16x32_bf16(false, ahi[s], false, bh, (short)0, c, false, false);
      c = __builtin_amdgcn_wmma_f32_16x16x32_bf16(false, ahi[s], false, bl, (short)0, c, false, false);
      c = __builtin_amdgcn_wmma_f32_16x16x32_bf16(false, alo[s], false, bh, (short)0, c, false, false);
    }

    int colg = jt * 16 + nrow;
#pragma unroll
    for (int r = 0; r < 8; ++r) {
      int rowg = crow0 + r;
      if (colg == rowg) continue;  // exclude diagonal
      float v = c[r] * INVT;
      if (v > mrow[r]) { srow[r] = srow[r] * __expf(mrow[r] - v) + 1.0f; mrow[r] = v; }
      else              srow[r] += __expf(v - mrow[r]);
    }
  }

  // merge (m,s) across the 16 lanes sharing each row group
#pragma unroll
  for (int off = 1; off < 16; off <<= 1) {
#pragma unroll
    for (int r = 0; r < 8; ++r) {
      float om = __shfl_xor(mrow[r], off, 16);
      float os = __shfl_xor(srow[r], off, 16);
      float nm = fmaxf(mrow[r], om);
      srow[r] = srow[r] * __expf(mrow[r] - nm) + os * __expf(om - nm);
      mrow[r] = nm;
    }
  }
  if ((lane & 15) == 0) {
#pragma unroll
    for (int r = 0; r < 8; ++r)
      lse[crow0 + r] = mrow[r] + __logf(srow[r]);
  }
}

// ---------------- kernel 2: base loss + class counts ----------------
__global__ void __launch_bounds__(256) base_kernel(const float* __restrict__ b,
                                                   const int* __restrict__ y,
                                                   const float* __restrict__ lse,
                                                   float* counts, float* out) {
  int i = blockIdx.x * 256 + threadIdx.x;
  if (i >= WN) return;
  int yi = y[i];
  atomicAdd(&counts[yi], 1.0f);
  int jt = -1;
  for (int j = 0; j < WN; ++j) {
    if (j != i && y[j] == yi) { jt = j; break; }
  }
  if (jt < 0) jt = (i == 0) ? 1 : 0;
  float dot = 0.0f;
  for (int k = 0; k < WBITS; ++k) dot += b[i * WBITS + k] * b[jt * WBITS + k];
  atomicAdd(out, (lse[i] - dot * INVT) * (1.0f / WN));
}

// ---------------- kernel 3: fused mapnet + mapLoss + mean/segment sums ----------------
__global__ void __launch_bounds__(256) mapnet_kernel(
    const float* __restrict__ z, const int* __restrict__ y,
    const float* __restrict__ W1, const float* __restrict__ b1,
    const __bf16* __restrict__ w2t, const float* __restrict__ b2,
    const int* __restrict__ perm,
    const unsigned char* __restrict__ maskMap,
    const unsigned char* __restrict__ maskOut,
    float* __restrict__ meanSum, float* __restrict__ seg, float* __restrict__ out) {
  __shared__ float W1s[8 * 256];
  __shared__ float b1s[256], b2s[256];
  __shared__ float xss[128 * 8];
  __shared__ int   tgts[128], clss[128];
  __shared__ float colsum[256];

  const int m = blockIdx.x;
  const int rowBase = blockIdx.y * 128;
  const int tid = threadIdx.x;

  for (int t = tid; t < 2048; t += 256) W1s[t] = W1[m * 2048 + t];
  b1s[tid] = b1[m * 256 + tid];
  b2s[tid] = b2[m * 256 + tid];
  colsum[tid] = 0.0f;
  if (tid < 128) {
    int rowg = rowBase + tid;
    clss[tid] = y[rowg];
    int tg = 0;
#pragma unroll
    for (int i2 = 0; i2 < 8; ++i2) {
      int bit = m * 8 + i2;
      int p = perm[bit];
      float zv = z[rowg * WBITS + p];
      float raw = maskOut[rowg * WBITS + bit] ? -zv : zv;
      if (raw > 0.0f) tg |= (1 << i2);
      xss[tid * 8 + i2] = maskMap[rowg * WBITS + bit] ? -zv : zv;
    }
    tgts[tid] = tg;
  }
  __syncthreads();

  const int lane = tid & 31;
  const int wave = tid >> 5;
  const int rt = wave * 16;
  const int nrow = lane & 15;
  const int ko = koff_of(lane);

  float xr[8];
#pragma unroll
  for (int i2 = 0; i2 < 8; ++i2) xr[i2] = xss[(rt + nrow) * 8 + i2];

  // A fragments = silu(xs@W1 + b1) for row rt+nrow, K=256 -> 8 slices
  v16bf af[8];
#pragma unroll
  for (int s = 0; s < 8; ++s) {
#pragma unroll
    for (int e = 0; e < 16; ++e) {
      int k = 32 * s + ko + ((e < 8) ? e : (e + 8));
      float acc = b1s[k];
#pragma unroll
      for (int i2 = 0; i2 < 8; ++i2) acc += xr[i2] * W1s[i2 * 256 + k];
      float h = acc / (1.0f + __expf(-acc));  // silu
      af[s][e] = (__bf16)h;
    }
  }

  const int crow0 = rt + ((lane < 16) ? 0 : 8);
  int clr[8], tgr[8];
#pragma unroll
  for (int r = 0; r < 8; ++r) { clr[r] = clss[crow0 + r]; tgr[r] = tgts[crow0 + r]; }

  float mrow[8], srow[8], tsum[8];
#pragma unroll
  for (int r = 0; r < 8; ++r) { mrow[r] = -1e30f; srow[r] = 0.0f; tsum[r] = 0.0f; }

  const __bf16* w2m = w2t + (size_t)m * 256 * 256;  // [o][k] bf16

  for (int ot = 0; ot < 16; ++ot) {
    int ocol = ot * 16 + nrow;
    if (ot + 1 < 16)  // warm L2/L0 for next o-tile's weights (global_prefetch_b8)
      __builtin_prefetch(w2m + (ocol + 16) * 256, 0, 3);
    v8f c = {};
#pragma unroll
    for (int s = 0; s < 8; ++s) {
      v16bf bf = frag_from(w2m + ocol * 256 + 32 * s + ko);  // B frag: 2x16B loads
      c = __builtin_amdgcn_wmma_f32_16x16x32_bf16(false, af[s], false, bf, (short)0, c, false, false);
    }
    float bias = b2s[ocol];
    float colp = 0.0f;
#pragma unroll
    for (int r = 0; r < 8; ++r) {
      float v = c[r] + bias;  // dec[row, m, ocol]
      if (v > mrow[r]) { srow[r] = srow[r] * __expf(mrow[r] - v) + 1.0f; mrow[r] = v; }
      else              srow[r] += __expf(v - mrow[r]);
      if (tgr[r] == ocol) tsum[r] = v;
      colp += v;
      atomicAdd(&seg[(size_t)clr[r] * (WM * 256) + m * 256 + ocol], v);
    }
    atomicAdd(&colsum[ocol], colp);
  }

#pragma unroll
  for (int off = 1; off < 16; off <<= 1) {
#pragma unroll
    for (int r = 0; r < 8; ++r) {
      float om = __shfl_xor(mrow[r], off, 16);
      float os = __shfl_xor(srow[r], off, 16);
      float ot2 = __shfl_xor(tsum[r], off, 16);
      float nm = fmaxf(mrow[r], om);
      srow[r] = srow[r] * __expf(mrow[r] - nm) + os * __expf(om - nm);
      mrow[r] = nm;
      tsum[r] += ot2;
    }
  }
  if ((lane & 15) == 0) {
    float part = 0.0f;
#pragma unroll
    for (int r = 0; r < 8; ++r)
      part += (mrow[r] + __logf(srow[r]) - tsum[r]);
    atomicAdd(out, part * (1.0f / WN));  // mapLoss
  }
  __syncthreads();
  atomicAdd(&meanSum[m * 256 + tid], colsum[tid]);
}

// ---------------- kernel 4: entropies -> netLoss ----------------
__device__ __forceinline__ float blkMax(float v, volatile float* sh) {
  int t = threadIdx.x;
  sh[t] = v; __syncthreads();
  for (int s = 128; s > 0; s >>= 1) { if (t < s) sh[t] = fmaxf((float)sh[t], (float)sh[t + s]); __syncthreads(); }
  float r = sh[0]; __syncthreads(); return r;
}
__device__ __forceinline__ float blkSum(float v, volatile float* sh) {
  int t = threadIdx.x;
  sh[t] = v; __syncthreads();
  for (int s = 128; s > 0; s >>= 1) { if (t < s) sh[t] = (float)sh[t] + (float)sh[t + s]; __syncthreads(); }
  float r = sh[0]; __syncthreads(); return r;
}

__global__ void __launch_bounds__(256) finale_kernel(const float* __restrict__ counts,
                                                     const float* __restrict__ meanSum,
                                                     const float* __restrict__ seg,
                                                     float* out) {
  __shared__ float sh[256];
  int t = threadIdx.x;
  if ((int)blockIdx.x < WC) {
    int cidx = blockIdx.x;
    float cnt = counts[cidx];
    if (cnt <= 0.0f) return;  // uniform exit
    float inv = 1.0f / cnt;
    float acc = 0.0f;
    for (int m = 0; m < WM; ++m) {
      float x = seg[(size_t)cidx * WM * 256 + m * 256 + t] * inv;
      float mx = blkMax(x, sh);
      float ex = __expf(x - mx);
      float S  = blkSum(ex, sh);
      float px = blkSum(ex * x, sh);
      acc += (mx + __logf(S)) - px / S;
    }
    if (t == 0) atomicAdd(out, LAMW * acc);  // + LAM * intra
  } else {
    float acc = 0.0f;
    for (int m = 0; m < WM; ++m) {
      float x = meanSum[m * 256 + t] * (1.0f / WN);
      float mx = blkMax(x, sh);
      float ex = __expf(x - mx);
      float S  = blkSum(ex, sh);
      float px = blkSum(ex * x, sh);
      acc += (mx + __logf(S)) - px / S;
    }
    if (t == 0) atomicAdd(out, -acc);  // - inter
  }
}

extern "C" void kernel_launch(void* const* d_in, const int* in_sizes, int n_in,
                              void* d_out, int out_size, void* d_ws, size_t ws_size,
                              hipStream_t stream) {
  const float* b  = (const float*)d_in[0];
  const float* z  = (const float*)d_in[1];
  const int*   y  = (const int*)d_in[2];
  const float* W1 = (const float*)d_in[3];
  const float* b1 = (const float*)d_in[4];
  const float* W2 = (const float*)d_in[5];
  const float* b2 = (const float*)d_in[6];
  const int*   perm = (const int*)d_in[7];
  const unsigned char* mm = (const unsigned char*)d_in[8];
  const unsigned char* mo = (const unsigned char*)d_in[9];
  float* out = (float*)d_out;
  float* ws  = (float*)d_ws;

  float*  lse     = ws + OFF_LSE;
  float*  counts  = ws + OFF_CNT;
  float*  meanSum = ws + OFF_MEAN;
  float*  seg     = ws + OFF_SEG;
  __bf16* bhi     = (__bf16*)(ws + OFF_BHI);
  __bf16* blo     = (__bf16*)(ws + OFF_BLO);
  __bf16* w2t     = (__bf16*)(ws + OFF_W2T);

  hipLaunchKernelGGL(init_kernel,    dim3(512),          dim3(256), 0, stream, ws, out);
  hipLaunchKernelGGL(prep_b_kernel,  dim3(WN*WBITS/256), dim3(256), 0, stream, b, bhi, blo);
  hipLaunchKernelGGL(prep_w2_kernel, dim3(WM*256),       dim3(256), 0, stream, W2, w2t);
  hipLaunchKernelGGL(lse_kernel,     dim3(WN / 128),     dim3(256), 0, stream, bhi, blo, lse);
  hipLaunchKernelGGL(base_kernel,    dim3(WN / 256),     dim3(256), 0, stream, b, y, lse, counts, out);
  hipLaunchKernelGGL(mapnet_kernel,  dim3(WM, WN / 128), dim3(256), 0, stream,
                     z, y, W1, b1, w2t, b2, perm, mm, mo, meanSum, seg, out);
  hipLaunchKernelGGL(finale_kernel,  dim3(WC + 1),       dim3(256), 0, stream,
                     counts, meanSum, seg, out);
}
